// BiMPM_Model_52776558133586
// MI455X (gfx1250) — matrix operand order repaired and checked by reading, not compiled, not run on hardware
//
#include <hip/hip_runtime.h>

#define BB 8
#define SS 192
#define HH 192
#define LL 20
#define EE 300
#define EP 320
#define GG 768    // 4H
#define DH 384    // 2H
#define MROWS 1536 // B*S
#define EPSV 1e-8f
#define NT48 48   // 768/16 n-tiles in packed weight layout

typedef __bf16 bf16_t;
typedef __attribute__((ext_vector_type(16))) __bf16 v16bf;
typedef __attribute__((ext_vector_type(8)))  float  v8f;

// ---------------------------------------------------------------- WMMA helpers
__device__ inline v8f wmma_bf16(v16bf a, v16bf b, v8f c) {
  return __builtin_amdgcn_wmma_f32_16x16x32_bf16(false, a, false, b, (short)0, c, false, false);
}

// A matrix 16x32 (MxK), row-major source. contiguous k order:
// lanes 0-15: M=lane, K 0..15; lanes 16-31: K 16..31.
// Per lane the 16 values form two contiguous 16B runs -> 2x ds_load_b128.
__device__ inline v16bf load_a_frag(const bf16_t* A, int lda, int row0, int k0) {
  int lane = threadIdx.x & 31;
  int m = lane & 15, half = lane >> 4;
  const bf16_t* base = A + (size_t)(row0 + m) * lda + k0;
  v16bf r;
#pragma unroll
  for (int v = 0; v < 8; ++v) {
    int k = 2 * v + 16 * half;
    r[2 * v]     = base[k];
    r[2 * v + 1] = base[k + 1];
  }
  return r;
}

// B operand stored in fragment order: region(kt, ntile) = 32 lanes x 16 bf16,
// contiguous per lane (k = 16*half + elem, same order as load_a_frag).
// lane = (k&31)>=16 ? 16+n : n ; elem = k&15.
__device__ inline v16bf load_b_packed(const bf16_t* Bp, int kt, int ntile, int nTiles) {
  int lane = threadIdx.x & 31;
  const bf16_t* p = Bp + ((size_t)(kt * nTiles + ntile) * 32 + lane) * 16;
  union { uint4 u[2]; v16bf v; } cv;
  cv.u[0] = ((const uint4*)p)[0];
  cv.u[1] = ((const uint4*)p)[1];
  return cv.v;
}

// C/D 16x16 f32: lane n = lane&15; row m = v + 8*(lane>>4)
__device__ inline void store_c(float* C, int ldc, int row0, int col0, v8f acc) {
  int lane = threadIdx.x & 31;
  int n = lane & 15, half = lane >> 4;
#pragma unroll
  for (int v = 0; v < 8; ++v)
    C[(size_t)(row0 + v + 8 * half) * ldc + col0 + n] = acc[v];
}

__device__ inline float sigf(float x) { return 1.f / (1.f + __expf(-x)); }

// monotonic float<->uint encode for atomic max of signed floats
__device__ inline unsigned fenc(float f) {
  unsigned u = __float_as_uint(f);
  return (u & 0x80000000u) ? ~u : (u | 0x80000000u);
}
__device__ inline float fdec(unsigned u) {
  unsigned v = (u & 0x80000000u) ? (u & 0x7fffffffu) : ~u;
  return __uint_as_float(v);
}

// LDS byte offset of a generic pointer to __shared__ (flat addr low 32 bits)
__device__ inline unsigned lds_off(const void* p) {
  return (unsigned)(unsigned long long)p;
}

// async memory->LDS copy of 16 bytes per lane (ASYNCcnt tracked)
__device__ inline void async_b128(unsigned ldsaddr, const void* gaddr) {
  asm volatile("global_load_async_to_lds_b128 %0, %1, off"
               :: "v"(ldsaddr), "v"(gaddr) : "memory");
}
__device__ inline void wait_async0() {
  asm volatile("s_wait_asynccnt 0" ::: "memory");
}

// ---------------------------------------------------------------- pack kernels
__global__ void k_embed(const int* __restrict__ text, const float* __restrict__ emb,
                        bf16_t* __restrict__ out) {
  int row = blockIdx.x;
  int idx = text[row];
  for (int k = threadIdx.x; k < EP; k += 256)
    out[(size_t)row * EP + k] = (k < EE) ? (bf16_t)emb[(size_t)idx * EE + k] : (bf16_t)0.f;
}

// W (768 x K) fp32 -> W^T (KP x 768) bf16 in WMMA fragment order (nTiles=48)
__global__ void k_wtpack(const float* __restrict__ W, bf16_t* __restrict__ out, int K, int KP) {
  int i = blockIdx.x * 256 + threadIdx.x;
  if (i >= KP * GG) return;
  int k = i / GG, n = i % GG;
  float v = (k < K) ? W[(size_t)n * K + k] : 0.f;
  int kt = k >> 5, kk = k & 31;
  int ntile = n >> 4, nn = n & 15;
  int lane = ((kk >> 4) << 4) + nn;
  int elem = kk & 15;
  out[((size_t)(kt * NT48 + ntile) * 32 + lane) * 16 + elem] = (bf16_t)v;
}

__global__ void k_mvpack(const float* __restrict__ mv, bf16_t* __restrict__ out) {
  int row = blockIdx.x;
  for (int k = threadIdx.x; k < HH; k += 256)
    out[(size_t)row * HH + k] = (k < 160) ? (bf16_t)mv[(size_t)row * 160 + k] : (bf16_t)0.f;
}

// ---------------------------------------------------------------- generic GEMM
// C[1536x768] = A[1536 x lda bf16] x Bpacked[K x 768]; grid (12,12), 256 thr.
// A tile staged via GLOBAL_LOAD_ASYNC_TO_LDS_B128; B frags direct from the
// fragment-packed (L2-resident) weight buffer as 2x global b128 per lane.
__global__ void k_gemm(const bf16_t* __restrict__ A, const bf16_t* __restrict__ Bp,
                       float* __restrict__ C, int lda, int kIters) {
  __shared__ __align__(16) bf16_t As[128 * 32];
  const int tid = threadIdx.x;
  const int w = tid >> 5;
  const int row0 = blockIdx.y * 128;
  const int ntile0 = blockIdx.x * 4;
  const unsigned as_base = lds_off(&As[0]);
  v8f acc[4];
  for (int j = 0; j < 4; ++j)
    for (int i = 0; i < 8; ++i) acc[j][i] = 0.f;
  for (int kt = 0; kt < kIters; ++kt) {
    int k0 = kt * 32;
#pragma unroll
    for (int i = 0; i < 2; ++i) {
      int lin = tid * 2 + i;           // 512 x 16B = 8KB tile
      int r = lin >> 2, q = lin & 3;   // 128 rows x 4 chunks
      const char* g = (const char*)(A + (size_t)(row0 + r) * lda + k0) + q * 16;
      async_b128(as_base + (unsigned)(r * 64 + q * 16), g);
    }
    wait_async0();
    __syncthreads();
    v16bf a = load_a_frag(As, 32, w * 16, 0);
#pragma unroll
    for (int j = 0; j < 4; ++j) {
      v16bf b = load_b_packed(Bp, kt, ntile0 + j, NT48);
      acc[j] = wmma_bf16(a, b, acc[j]);
    }
    __syncthreads();
  }
  for (int j = 0; j < 4; ++j)
    store_c(C, GG, row0 + w * 16, (ntile0 + j) * 16, acc[j]);
}

// ---------------------------------------------------------------- LSTM scan
struct ScanArgs {
  const float*  gx[4];     // [B*S][768] precomputed x@Wih^T
  const bf16_t* whh[4];    // Whh^T fragment-packed [192x768]
  const float*  bias[4];   // [768]
  float*        outh[4];   // per-step h out [B][S][384] at outcol, or null
  float*        outfin[4]; // final h [B][192], or null
  int           outcol[4];
  int           rev[4];
};

__global__ void k_lstm(ScanArgs sa) {
  __shared__ __align__(16) bf16_t hbf[16 * HH]; // padded M=16 A operand
  __shared__ float gbuf[8 * GG];
  __shared__ float cbuf[8 * HH];
  __shared__ float hfl[8 * HH];
  const int id = blockIdx.x;
  const float*  gx   = sa.gx[id];
  const bf16_t* whh  = sa.whh[id];
  const float*  bias = sa.bias[id];
  float* outh   = sa.outh[id];
  float* outfin = sa.outfin[id];
  const int outcol = sa.outcol[id];
  const int rev    = sa.rev[id];
  const int tid = threadIdx.x;
  const int w = tid >> 5, lane = tid & 31;
  const int n_lane = lane & 15, half = lane >> 4;

  // warm Whh^T into cache before the serial scan (global_prefetch_b8)
  for (int i = tid; i < (HH * GG * 2) / 256; i += 256)
    __builtin_prefetch((const char*)whh + (size_t)i * 256, 0, 3);

  for (int i = tid; i < 16 * HH; i += 256) hbf[i] = (bf16_t)0.f;
  for (int i = tid; i < 8 * HH; i += 256) { cbuf[i] = 0.f; hfl[i] = 0.f; }
  __syncthreads();

  for (int t = 0; t < SS; ++t) {
    int ta = rev ? (SS - 1 - t) : t;
    v8f acc[6];
    for (int j = 0; j < 6; ++j)
      for (int i = 0; i < 8; ++i) acc[j][i] = 0.f;
    for (int kk = 0; kk < 6; ++kk) {
      v16bf a = load_a_frag(hbf, HH, 0, kk * 32);
#pragma unroll
      for (int j = 0; j < 6; ++j) {
        v16bf b = load_b_packed(whh, kk, w * 6 + j, NT48);
        acc[j] = wmma_bf16(a, b, acc[j]);
      }
    }
#pragma unroll
    for (int j = 0; j < 6; ++j) {
      int col = w * 96 + j * 16 + n_lane;
#pragma unroll
      for (int v = 0; v < 8; ++v) {
        int m = v + 8 * half;
        if (m < 8)
          gbuf[m * GG + col] = acc[j][v] + gx[(size_t)(m * SS + ta) * GG + col] + bias[col];
      }
    }
    __syncthreads();
    for (int idx = tid; idx < 8 * HH; idx += 256) {
      int b = idx / HH, jj = idx % HH;
      float gi = gbuf[b * GG + jj];
      float gf = gbuf[b * GG + HH + jj];
      float gg = gbuf[b * GG + 2 * HH + jj];
      float go = gbuf[b * GG + 3 * HH + jj];
      float c = sigf(gf) * cbuf[idx] + sigf(gi) * tanhf(gg);
      float h = sigf(go) * tanhf(c);
      cbuf[idx] = c; hfl[idx] = h; hbf[b * HH + jj] = (bf16_t)h;
      if (outh) outh[(size_t)(b * SS + ta) * DH + outcol + jj] = h;
    }
    __syncthreads();
  }
  if (outfin)
    for (int idx = tid; idx < 8 * HH; idx += 256) outfin[idx] = hfl[idx];
}

// ---------------------------------------------------------------- attention (per b)
// cos(p_s, h_t) -> att[b,s,t], row sums, col sums. B operand staged into LDS
// in fragment order (12 t-tiles) -> 2x ds_load_b128 per WMMA fragment.
__global__ void k_attn(const float* __restrict__ repP, const float* __restrict__ repH, int off,
                       float* __restrict__ att, float* __restrict__ rsumG, float* __restrict__ csumG) {
  extern __shared__ char smem[];
  bf16_t* Ab = (bf16_t*)smem;            // [s][k] row-major
  bf16_t* Bb = Ab + SS * HH;             // fragment order, 12 t-tiles
  float* n1 = (float*)(Bb + SS * HH);
  float* n2 = n1 + SS;
  float* rs = n2 + SS;
  float* cs = rs + SS;
  const int b = blockIdx.x;
  const int tid = threadIdx.x;
  for (int i = tid; i < SS * HH; i += 256) {
    int s = i / HH, k = i % HH;
    Ab[s * HH + k] = (bf16_t)repP[(size_t)(b * SS + s) * DH + off + k];
  }
  for (int i = tid; i < SS * HH; i += 256) {
    int t = i / HH, k = i % HH;
    float v = repH[(size_t)(b * SS + t) * DH + off + k];
    int kt = k >> 5, kk = k & 31, tt = t >> 4, tn = t & 15;
    Bb[((size_t)(kt * 12 + tt) * 32 + ((kk >> 4) << 4) + tn) * 16 + (kk & 15)] = (bf16_t)v;
  }
  if (tid < SS) { rs[tid] = 0.f; cs[tid] = 0.f; }
  __syncthreads();
  if (tid < SS) {
    float s1 = 0.f, s2 = 0.f;
    for (int k = 0; k < HH; ++k) { float v = (float)Ab[tid * HH + k]; s1 += v * v; }
    for (int k = 0; k < HH; ++k) { float v = repH[(size_t)(b * SS + tid) * DH + off + k]; s2 += v * v; }
    n1[tid] = sqrtf(s1); n2[tid] = sqrtf(s2);
  }
  __syncthreads();
  const int w = tid >> 5, lane = tid & 31;
  const int n_lane = lane & 15, half = lane >> 4;
  for (int i = 0; i < 18; ++i) {
    int tt = w * 18 + i;
    int s0 = (tt / 12) * 16, tc = tt % 12;
    v8f acc; for (int q = 0; q < 8; ++q) acc[q] = 0.f;
    for (int kk = 0; kk < 6; ++kk) {
      v16bf a = load_a_frag(Ab, HH, s0, kk * 32);
      v16bf bb = load_b_packed(Bb, kk, tc, 12);
      acc = wmma_bf16(a, bb, acc);
    }
    int tcol = tc * 16 + n_lane;
#pragma unroll
    for (int v = 0; v < 8; ++v) {
      int srow = s0 + v + 8 * half;
      float den = n1[srow] * n2[tcol];
      float cosv = acc[v] / (den > EPSV ? den : EPSV);
      att[(size_t)(b * SS + srow) * SS + tcol] = cosv;
      atomicAdd(&rs[srow], cosv);
      atomicAdd(&cs[tcol], cosv);
    }
  }
  __syncthreads();
  if (tid < SS) { rsumG[b * SS + tid] = rs[tid]; csumG[b * SS + tid] = cs[tid]; }
}

// ---------------------------------------------------------------- pairwise match (per b,l)
__global__ void k_pair(const float* __restrict__ repP, const float* __restrict__ repH, int off,
                       const float* __restrict__ wpair, float* __restrict__ mvp,
                       float* __restrict__ mvh, int colbase) {
  extern __shared__ char smem[];
  bf16_t* Ab = (bf16_t*)smem;
  bf16_t* Bb = Ab + SS * HH;             // fragment order, 12 t-tiles
  float* n1 = (float*)(Bb + SS * HH);
  float* n2 = n1 + SS;
  unsigned* rmax = (unsigned*)(n2 + SS);
  unsigned* cmax = rmax + SS;
  const int b = blockIdx.x / LL;
  const int l = blockIdx.x % LL;
  const float* wl = wpair + (size_t)l * HH;
  const int tid = threadIdx.x;
  for (int i = tid; i < SS * HH; i += 256) {
    int s = i / HH, k = i % HH;
    Ab[s * HH + k] = (bf16_t)(repP[(size_t)(b * SS + s) * DH + off + k] * wl[k]);
  }
  for (int i = tid; i < SS * HH; i += 256) {
    int t = i / HH, k = i % HH;
    float v = repH[(size_t)(b * SS + t) * DH + off + k] * wl[k];
    int kt = k >> 5, kk = k & 31, tt = t >> 4, tn = t & 15;
    Bb[((size_t)(kt * 12 + tt) * 32 + ((kk >> 4) << 4) + tn) * 16 + (kk & 15)] = (bf16_t)v;
  }
  if (tid < SS) { rmax[tid] = 0u; cmax[tid] = 0u; }
  __syncthreads();
  if (tid < SS) {
    float s1 = 0.f, s2 = 0.f;
    for (int k = 0; k < HH; ++k) { float v = (float)Ab[tid * HH + k]; s1 += v * v; }
    for (int k = 0; k < HH; ++k) {
      float v = repH[(size_t)(b * SS + tid) * DH + off + k] * wl[k];
      s2 += v * v;
    }
    n1[tid] = sqrtf(s1); n2[tid] = sqrtf(s2);
  }
  __syncthreads();
  const int w = tid >> 5, lane = tid & 31;
  const int n_lane = lane & 15, half = lane >> 4;
  for (int i = 0; i < 18; ++i) {
    int tt = w * 18 + i;
    int s0 = (tt / 12) * 16, tc = tt % 12;
    v8f acc; for (int q = 0; q < 8; ++q) acc[q] = 0.f;
    for (int kk = 0; kk < 6; ++kk) {
      v16bf a = load_a_frag(Ab, HH, s0, kk * 32);
      v16bf bb = load_b_packed(Bb, kk, tc, 12);
      acc = wmma_bf16(a, bb, acc);
    }
    int tcol = tc * 16 + n_lane;
#pragma unroll
    for (int v = 0; v < 8; ++v) {
      int srow = s0 + v + 8 * half;
      float den = n1[srow] * n2[tcol];
      float cosv = acc[v] / (den > EPSV ? den : EPSV);
      unsigned e = fenc(cosv);
      atomicMax(&rmax[srow], e);
      atomicMax(&cmax[tcol], e);
    }
  }
  __syncthreads();
  if (tid < SS) {
    mvp[(size_t)(b * SS + tid) * 160 + colbase + 20 + l] = fdec(rmax[tid]);
    mvh[(size_t)(b * SS + tid) * 160 + colbase + 20 + l] = fdec(cmax[tid]);
  }
}

// ---------------------------------------------------------------- attention mean/max
__global__ void k_meanmax(const float* __restrict__ att, const float* __restrict__ rsum,
                          const float* __restrict__ csum, const float* __restrict__ repP,
                          const float* __restrict__ repH, int off,
                          float* __restrict__ mean_h, float* __restrict__ maxa_h,
                          float* __restrict__ mean_p, float* __restrict__ maxa_p) {
  __shared__ float arow[SS];
  const int bs = blockIdx.x;
  const int b = bs / SS, s = bs % SS;
  const int k = threadIdx.x; // 192 threads
  if (blockIdx.y == 0) {
    arow[k] = att[(size_t)bs * SS + k];
    __syncthreads();
    float acc = 0.f, mx = -3.4e38f;
    for (int t = 0; t < SS; ++t) {
      float a = arow[t];
      float hv = repH[(size_t)(b * SS + t) * DH + off + k];
      acc += a * hv; mx = fmaxf(mx, a * hv);
    }
    float dd = rsum[bs]; dd = dd > EPSV ? dd : EPSV;
    mean_h[(size_t)bs * HH + k] = acc / dd;
    maxa_h[(size_t)bs * HH + k] = mx;
  } else {
    arow[k] = att[(size_t)(b * SS + k) * SS + s];
    __syncthreads();
    float acc = 0.f, mx = -3.4e38f;
    for (int t = 0; t < SS; ++t) {
      float a = arow[t];
      float pv = repP[(size_t)(b * SS + t) * DH + off + k];
      acc += a * pv; mx = fmaxf(mx, a * pv);
    }
    float dd = csum[bs]; dd = dd > EPSV ? dd : EPSV;
    mean_p[(size_t)bs * HH + k] = acc / dd;
    maxa_p[(size_t)bs * HH + k] = mx;
  }
}

// ---------------------------------------------------------------- full/mean/maxa cosines
__global__ void k_fullcos(const float* __restrict__ repP, const float* __restrict__ repH, int off,
                          const float* __restrict__ wfull, const float* __restrict__ wmean,
                          const float* __restrict__ wmax,
                          const float* __restrict__ mean_h, const float* __restrict__ maxa_h,
                          const float* __restrict__ mean_p, const float* __restrict__ maxa_p,
                          int idxlast, float* __restrict__ mvp, float* __restrict__ mvh,
                          int colbase) {
  __shared__ float v1s[HH], pf[HH], pm[HH], px[HH];
  const int bs = blockIdx.x;
  const int b = bs / SS;
  const int side = blockIdx.y;
  const int tid = threadIdx.x;
  for (int k = tid; k < HH; k += 256) {
    if (side == 0) {
      v1s[k] = repP[(size_t)bs * DH + off + k];
      pf[k]  = repH[(size_t)(b * SS + idxlast) * DH + off + k];
      pm[k]  = mean_h[(size_t)bs * HH + k];
      px[k]  = maxa_h[(size_t)bs * HH + k];
    } else {
      v1s[k] = repH[(size_t)bs * DH + off + k];
      pf[k]  = repP[(size_t)(b * SS + idxlast) * DH + off + k];
      pm[k]  = mean_p[(size_t)bs * HH + k];
      px[k]  = maxa_p[(size_t)bs * HH + k];
    }
  }
  __syncthreads();
  const int w = tid >> 5, lane = tid & 31;
  float* outp = (side == 0) ? mvp : mvh;
  for (int which = 0; which < 3; ++which) {
    const float* ww = (which == 0) ? wfull : (which == 1 ? wmean : wmax);
    const float* v2 = (which == 0) ? pf : (which == 1 ? pm : px);
    for (int li = 0; li < 3; ++li) {
      int l = w + 8 * li;
      if (l >= LL) continue;
      float sab = 0.f, saa = 0.f, sbb = 0.f;
      for (int k = lane; k < HH; k += 32) {
        float wv = ww[l * HH + k];
        float a = v1s[k] * wv, bv = v2[k] * wv;
        sab += a * bv; saa += a * a; sbb += bv * bv;
      }
      for (int o = 16; o > 0; o >>= 1) {
        sab += __shfl_down(sab, o, 32);
        saa += __shfl_down(saa, o, 32);
        sbb += __shfl_down(sbb, o, 32);
      }
      if (lane == 0) {
        float na = sqrtf(saa); na = na > EPSV ? na : EPSV;
        float nb = sqrtf(sbb); nb = nb > EPSV ? nb : EPSV;
        int woff = (which == 0) ? 0 : (which == 1 ? 40 : 60);
        outp[(size_t)bs * 160 + colbase + woff + l] = sab / (na * nb);
      }
    }
  }
}

// ---------------------------------------------------------------- classifier
__global__ void k_cls(const float* hpf, const float* hpb, const float* hhf, const float* hhb,
                      const float* w1, const float* b1, const float* w2, const float* b2,
                      float* out) {
  __shared__ float xv[768];
  __shared__ float y1[384];
  __shared__ float red[256];
  const int b = blockIdx.x, tid = threadIdx.x;
  for (int j = tid; j < 768; j += 256) {
    const float* src = (j < 192) ? hpf : (j < 384) ? hpb : (j < 576) ? hhf : hhb;
    xv[j] = src[b * 192 + (j % 192)];
  }
  __syncthreads();
  for (int j = tid; j < 384; j += 256) {
    float acc = b1[j];
    for (int k = 0; k < 768; ++k) acc += w1[(size_t)j * 768 + k] * xv[k];
    y1[j] = tanhf(acc);
  }
  __syncthreads();
  for (int o = 0; o < 2; ++o) {
    float p = 0.f;
    for (int j = tid; j < 384; j += 256) p += w2[o * 384 + j] * y1[j];
    red[tid] = p; __syncthreads();
    for (int st = 128; st > 0; st >>= 1) { if (tid < st) red[tid] += red[tid + st]; __syncthreads(); }
    if (tid == 0) out[b * 2 + o] = red[0] + b2[o];
    __syncthreads();
  }
}

// ---------------------------------------------------------------- host driver
extern "C" void kernel_launch(void* const* d_in, const int* in_sizes, int n_in,
                              void* d_out, int out_size, void* d_ws, size_t ws_size,
                              hipStream_t stream) {
  const int*   text_a    = (const int*)d_in[0];
  const int*   text_b    = (const int*)d_in[1];
  const float* emb       = (const float*)d_in[2];
  const float* ctx_Wih_f = (const float*)d_in[3];
  const float* ctx_Whh_f = (const float*)d_in[4];
  const float* ctx_b_f   = (const float*)d_in[5];
  const float* ctx_Wih_b = (const float*)d_in[6];
  const float* ctx_Whh_b = (const float*)d_in[7];
  const float* ctx_b_b   = (const float*)d_in[8];
  const float* agg_Wih_f = (const float*)d_in[9];
  const float* agg_Whh_f = (const float*)d_in[10];
  const float* agg_b_f   = (const float*)d_in[11];
  const float* agg_Wih_b = (const float*)d_in[12];
  const float* agg_Whh_b = (const float*)d_in[13];
  const float* agg_b_b   = (const float*)d_in[14];
  const float* mp_w      = (const float*)d_in[15];
  const float* fc1_w     = (const float*)d_in[16];
  const float* fc1_b     = (const float*)d_in[17];
  const float* fc2_w     = (const float*)d_in[18];
  const float* fc2_b     = (const float*)d_in[19];
  float* out = (float*)d_out;
  (void)in_sizes; (void)n_in; (void)out_size; (void)ws_size;

  char* ws = (char*)d_ws;
  size_t cur = 0;
  auto alloc = [&](size_t bytes) -> char* {
    char* p = ws + cur;
    cur = (cur + bytes + 255) & ~(size_t)255;
    return p;
  };

  bf16_t* xa = (bf16_t*)alloc((size_t)MROWS * EP * 2);
  bf16_t* xb = (bf16_t*)alloc((size_t)MROWS * EP * 2);
  bf16_t* wih_cf = (bf16_t*)alloc((size_t)EP * GG * 2);
  bf16_t* wih_cb = (bf16_t*)alloc((size_t)EP * GG * 2);
  bf16_t* whh_cf = (bf16_t*)alloc((size_t)HH * GG * 2);
  bf16_t* whh_cb = (bf16_t*)alloc((size_t)HH * GG * 2);
  bf16_t* wih_af = (bf16_t*)alloc((size_t)HH * GG * 2);
  bf16_t* wih_ab = (bf16_t*)alloc((size_t)HH * GG * 2);
  bf16_t* whh_af = (bf16_t*)alloc((size_t)HH * GG * 2);
  bf16_t* whh_ab = (bf16_t*)alloc((size_t)HH * GG * 2);
  float* gates0 = (float*)alloc((size_t)MROWS * GG * 4);
  float* gates1 = (float*)alloc((size_t)MROWS * GG * 4);
  float* gates2 = (float*)alloc((size_t)MROWS * GG * 4);
  float* gates3 = (float*)alloc((size_t)MROWS * GG * 4);
  float* repq = (float*)alloc((size_t)MROWS * DH * 4);
  float* repd = (float*)alloc((size_t)MROWS * DH * 4);
  float* att  = (float*)alloc((size_t)MROWS * SS * 4);
  float* rsum = (float*)alloc((size_t)MROWS * 4);
  float* csum = (float*)alloc((size_t)MROWS * 4);
  float* mean_h = (float*)alloc((size_t)MROWS * HH * 4);
  float* maxa_h = (float*)alloc((size_t)MROWS * HH * 4);
  float* mean_p = (float*)alloc((size_t)MROWS * HH * 4);
  float* maxa_p = (float*)alloc((size_t)MROWS * HH * 4);
  float* mvp = (float*)alloc((size_t)MROWS * 160 * 4);
  float* mvh = (float*)alloc((size_t)MROWS * 160 * 4);
  bf16_t* mvp_bf = (bf16_t*)alloc((size_t)MROWS * HH * 2);
  bf16_t* mvh_bf = (bf16_t*)alloc((size_t)MROWS * HH * 2);
  float* hpf = (float*)alloc(BB * HH * 4);
  float* hpb = (float*)alloc(BB * HH * 4);
  float* hhf = (float*)alloc(BB * HH * 4);
  float* hhb = (float*)alloc(BB * HH * 4);

  // 1) embedding gather + pad to K=320 bf16
  k_embed<<<MROWS, 256, 0, stream>>>(text_a, emb, xa);
  k_embed<<<MROWS, 256, 0, stream>>>(text_b, emb, xb);

  // 2) weight transposes -> fragment-packed bf16
  int nb;
  nb = (EP * GG + 255) / 256;
  k_wtpack<<<nb, 256, 0, stream>>>(ctx_Wih_f, wih_cf, EE, EP);
  k_wtpack<<<nb, 256, 0, stream>>>(ctx_Wih_b, wih_cb, EE, EP);
  nb = (HH * GG + 255) / 256;
  k_wtpack<<<nb, 256, 0, stream>>>(ctx_Whh_f, whh_cf, HH, HH);
  k_wtpack<<<nb, 256, 0, stream>>>(ctx_Whh_b, whh_cb, HH, HH);
  k_wtpack<<<nb, 256, 0, stream>>>(agg_Wih_f, wih_af, 160, HH);
  k_wtpack<<<nb, 256, 0, stream>>>(agg_Wih_b, wih_ab, 160, HH);
  k_wtpack<<<nb, 256, 0, stream>>>(agg_Whh_f, whh_af, HH, HH);
  k_wtpack<<<nb, 256, 0, stream>>>(agg_Whh_b, whh_ab, HH, HH);

  // 3) input-projection GEMMs (gates_x = x @ Wih^T)
  dim3 gg(12, 12);
  k_gemm<<<gg, 256, 0, stream>>>(xa, wih_cf, gates0, EP, 10);
  k_gemm<<<gg, 256, 0, stream>>>(xa, wih_cb, gates1, EP, 10);
  k_gemm<<<gg, 256, 0, stream>>>(xb, wih_cf, gates2, EP, 10);
  k_gemm<<<gg, 256, 0, stream>>>(xb, wih_cb, gates3, EP, 10);

  // 4) context BiLSTM scans (4 independent scans, one workgroup each)
  {
    ScanArgs sa{};
    sa.gx[0] = gates0; sa.whh[0] = whh_cf; sa.bias[0] = ctx_b_f; sa.outh[0] = repq; sa.outfin[0] = nullptr; sa.outcol[0] = 0;   sa.rev[0] = 0;
    sa.gx[1] = gates1; sa.whh[1] = whh_cb; sa.bias[1] = ctx_b_b; sa.outh[1] = repq; sa.outfin[1] = nullptr; sa.outcol[1] = HH;  sa.rev[1] = 1;
    sa.gx[2] = gates2; sa.whh[2] = whh_cf; sa.bias[2] = ctx_b_f; sa.outh[2] = repd; sa.outfin[2] = nullptr; sa.outcol[2] = 0;   sa.rev[2] = 0;
    sa.gx[3] = gates3; sa.whh[3] = whh_cb; sa.bias[3] = ctx_b_b; sa.outh[3] = repd; sa.outfin[3] = nullptr; sa.outcol[3] = HH;  sa.rev[3] = 1;
    k_lstm<<<4, 256, 0, stream>>>(sa);
  }

  // 5) matching (both directions)
  const size_t dynA = (size_t)2 * SS * HH * 2 + 4 * SS * 4; // attn LDS
  const size_t dynP = (size_t)2 * SS * HH * 2 + 4 * SS * 4; // pair LDS
  for (int d = 0; d < 2; ++d) {
    int off = d * HH;
    int colbase = d * 80;
    int idxlast = d ? 0 : (SS - 1);
    const float* wfull = mp_w + (size_t)(0 + d) * LL * HH;
    const float* wpair = mp_w + (size_t)(2 + d) * LL * HH;
    const float* wmean = mp_w + (size_t)(4 + d) * LL * HH;
    const float* wmaxw = mp_w + (size_t)(6 + d) * LL * HH;
    k_attn<<<BB, 256, dynA, stream>>>(repq, repd, off, att, rsum, csum);
    k_meanmax<<<dim3(MROWS, 2), HH, 0, stream>>>(att, rsum, csum, repq, repd, off,
                                                 mean_h, maxa_h, mean_p, maxa_p);
    k_pair<<<BB * LL, 256, dynP, stream>>>(repq, repd, off, wpair, mvp, mvh, colbase);
    k_fullcos<<<dim3(MROWS, 2), 256, 0, stream>>>(repq, repd, off, wfull, wmean, wmaxw,
                                                  mean_h, maxa_h, mean_p, maxa_p,
                                                  idxlast, mvp, mvh, colbase);
  }

  // 6) pack match vectors to bf16 (pad K 160->192)
  k_mvpack<<<MROWS, 256, 0, stream>>>(mvp, mvp_bf);
  k_mvpack<<<MROWS, 256, 0, stream>>>(mvh, mvh_bf);

  // 7) aggregation input projections
  k_gemm<<<gg, 256, 0, stream>>>(mvp_bf, wih_af, gates0, HH, 6);
  k_gemm<<<gg, 256, 0, stream>>>(mvp_bf, wih_ab, gates1, HH, 6);
  k_gemm<<<gg, 256, 0, stream>>>(mvh_bf, wih_af, gates2, HH, 6);
  k_gemm<<<gg, 256, 0, stream>>>(mvh_bf, wih_ab, gates3, HH, 6);

  // 8) aggregation BiLSTM scans (final hidden states only)
  {
    ScanArgs sa{};
    sa.gx[0] = gates0; sa.whh[0] = whh_af; sa.bias[0] = agg_b_f; sa.outh[0] = nullptr; sa.outfin[0] = hpf; sa.outcol[0] = 0; sa.rev[0] = 0;
    sa.gx[1] = gates1; sa.whh[1] = whh_ab; sa.bias[1] = agg_b_b; sa.outh[1] = nullptr; sa.outfin[1] = hpb; sa.outcol[1] = 0; sa.rev[1] = 1;
    sa.gx[2] = gates2; sa.whh[2] = whh_af; sa.bias[2] = agg_b_f; sa.outh[2] = nullptr; sa.outfin[2] = hhf; sa.outcol[2] = 0; sa.rev[2] = 0;
    sa.gx[3] = gates3; sa.whh[3] = whh_ab; sa.bias[3] = agg_b_b; sa.outh[3] = nullptr; sa.outfin[3] = hhb; sa.outcol[3] = 0; sa.rev[3] = 1;
    k_lstm<<<4, 256, 0, stream>>>(sa);
  }

  // 9) classifier head
  k_cls<<<BB, 256, 0, stream>>>(hpf, hpb, hhf, hhb, fc1_w, fc1_b, fc2_w, fc2_b, out);
}
